// TestTreeSelfAttention_30116310680246
// MI455X (gfx1250) — matrix operand order repaired and checked by reading, not compiled, not run on hardware
//
#include <hip/hip_runtime.h>
#include <cstdint>
#include <cstddef>

#define BB   2
#define SEQ  2048
#define DIM  512
#define NH   8
#define HD   64
#define RANK 128
#define DFF  1024
#define SP   (SEQ + 16)          // padded stride for V^T (zero tail)
#define EPSV 1e-6f

typedef __attribute__((ext_vector_type(16))) _Float16 v16h;
typedef __attribute__((ext_vector_type(8)))  _Float16 v8h;
typedef __attribute__((ext_vector_type(8)))  float    v8f;

__device__ __forceinline__ float sigmoidf_(float x){ return 1.f/(1.f+__expf(-x)); }

// ---------------------------------------------------------------------------
// fp32 -> f16 flat convert
__global__ void tts_f2h(const float* __restrict__ in, _Float16* __restrict__ out, size_t n){
  size_t i = (size_t)blockIdx.x*256 + threadIdx.x;
  if (i < n) out[i] = (_Float16)in[i];
}

// tril mask + convert for lat_w [SEQ,SEQ]
__global__ void tts_tril_f2h(const float* __restrict__ in, _Float16* __restrict__ out){
  size_t i = (size_t)blockIdx.x*256 + threadIdx.x;
  if (i >= (size_t)SEQ*SEQ) return;
  int t = (int)(i / SEQ), s = (int)(i % SEQ);
  out[i] = (s <= t) ? (_Float16)in[i] : (_Float16)0.f;
}

// batched transpose-convert: in [batch,R,C] f32 -> out [batch,C,R] f16
__global__ void tts_transpose_f2h(const float* __restrict__ in, _Float16* __restrict__ out,
                                  int R, int C){
  int b = blockIdx.y;
  int idx = blockIdx.x*256 + threadIdx.x;
  if (idx >= R*C) return;
  int r = idx / C, c = idx % C;
  out[(size_t)b*R*C + (size_t)c*R + r] = (_Float16)in[(size_t)b*R*C + (size_t)r*C + c];
}

// ---------------------------------------------------------------------------
// row RMSNorm (D=512), optional f32 and f16 outputs
__global__ void tts_rmsnorm(const float* __restrict__ x, const float* __restrict__ w,
                            float* __restrict__ outf, _Float16* __restrict__ outh){
  size_t row = blockIdx.x;
  const float* xr = x + row*DIM;
  __shared__ float red[256];
  float ss = 0.f;
  for (int i = threadIdx.x; i < DIM; i += 256){ float v = xr[i]; ss += v*v; }
  red[threadIdx.x] = ss; __syncthreads();
  for (int s = 128; s > 0; s >>= 1){
    if (threadIdx.x < s) red[threadIdx.x] += red[threadIdx.x+s];
    __syncthreads();
  }
  float r = rsqrtf(red[0]*(1.f/DIM) + EPSV);
  for (int i = threadIdx.x; i < DIM; i += 256){
    float v = xr[i]*r*w[i];
    if (outf) outf[row*DIM+i] = v;
    if (outh) outh[row*DIM+i] = (_Float16)v;
  }
}

// ---------------------------------------------------------------------------
// Register-blocked WMMA GEMM: each wave computes a 32x32 output block
// (2x2 tiles of 16x16 -> 4 WMMAs per load group, 2x data reuse).
// C[M,N](f32) = A[M,K](f16,row) x Bt[N,K](f16,row)^T. N must be mult of 32.
// act==1: sigmoid.  addsrc: residual added after act.  C / Ch nullable.
__global__ void __launch_bounds__(256)
tts_gemm_nt(const _Float16* __restrict__ A, const _Float16* __restrict__ Bt,
            float* __restrict__ C, _Float16* __restrict__ Ch,
            const float* __restrict__ addsrc, int M, int N, int K, int act){
  const int lane = threadIdx.x;                       // 0..31
  const int n0 = (blockIdx.x*blockDim.y + threadIdx.y)*32;
  const int m0 = blockIdx.y*32;
  if (n0 >= N) return;
  const int hg = lane >> 4, lr = lane & 15;
  int ar0 = m0 + lr;      if (ar0 >= M) ar0 = M - 1; // clamp; tails masked at store
  int ar1 = m0 + 16 + lr; if (ar1 >= M) ar1 = M - 1;
  const _Float16* Ap0 = A  + (size_t)ar0          * K;
  const _Float16* Ap1 = A  + (size_t)ar1          * K;
  const _Float16* Bp0 = Bt + (size_t)(n0 + lr)    * K;
  const _Float16* Bp1 = Bt + (size_t)(n0 + 16+lr) * K;
  v8f acc00 = {}, acc01 = {}, acc10 = {}, acc11 = {};
  for (int k0 = 0; k0 < K; k0 += 32){
    v8h x0 = *(const v8h*)(Ap0 + k0 + hg*16);
    v8h x1 = *(const v8h*)(Ap0 + k0 + hg*16 + 8);
    v8h y0 = *(const v8h*)(Ap1 + k0 + hg*16);
    v8h y1 = *(const v8h*)(Ap1 + k0 + hg*16 + 8);
    v16h b0 = *(const v16h*)(Bp0 + k0 + hg*16);
    v16h b1 = *(const v16h*)(Bp1 + k0 + hg*16);
    v16h a0f, a1f;
    #pragma unroll
    for (int e = 0; e < 8; ++e){
      a0f[e] = x0[e]; a0f[8+e] = x1[e];
      a1f[e] = y0[e]; a1f[8+e] = y1[e];
    }
    __builtin_prefetch((const void*)(Ap0 + k0 + 256), 0, 1);
    __builtin_prefetch((const void*)(Ap1 + k0 + 256), 0, 1);
    __builtin_prefetch((const void*)(Bp0 + k0 + 256), 0, 1);
    __builtin_prefetch((const void*)(Bp1 + k0 + 256), 0, 1);
    acc00 = __builtin_amdgcn_wmma_f32_16x16x32_f16(false, a0f, false, b0, (short)0, acc00, false, false);
    acc01 = __builtin_amdgcn_wmma_f32_16x16x32_f16(false, a0f, false, b1, (short)0, acc01, false, false);
    acc10 = __builtin_amdgcn_wmma_f32_16x16x32_f16(false, a1f, false, b0, (short)0, acc10, false, false);
    acc11 = __builtin_amdgcn_wmma_f32_16x16x32_f16(false, a1f, false, b1, (short)0, acc11, false, false);
  }
  auto emit = [&](int m, int n, float v){
    if (act == 1) v = sigmoidf_(v);
    if (addsrc) v += addsrc[(size_t)m*N + n];
    if (C)  C [(size_t)m*N + n] = v;
    if (Ch) Ch[(size_t)m*N + n] = (_Float16)v;
  };
  #pragma unroll
  for (int j = 0; j < 8; ++j){
    int mA = m0 + j + 8*hg;                           // C layout: M=j+8*hg, N=lr
    int mB = mA + 16;
    int nA = n0 + lr, nB = n0 + 16 + lr;
    if (mA < M){ emit(mA, nA, acc00[j]); emit(mA, nB, acc01[j]); }
    if (mB < M){ emit(mB, nA, acc10[j]); emit(mB, nB, acc11[j]); }
  }
}

// ---------------------------------------------------------------------------
// RoPE + repack [B,S,D] f32 -> [B,H,S,HD] f16 for q and k
__global__ void tts_rope_pack(const float* __restrict__ qf, const float* __restrict__ kf,
                              _Float16* __restrict__ qh, _Float16* __restrict__ kh){
  int row = blockIdx.x;                 // b*SEQ+s
  int b = row / SEQ, s = row % SEQ;
  int dg = threadIdx.x;                 // 0..511
  int h = dg >> 6, d = dg & 63, i = d & 31;
  float ang = (float)s * __expf(-(float)i * 0.28782313662f);  // ln(10000)/32
  float sn, cs; __sincosf(ang, &sn, &cs);
  size_t ib = (size_t)row*DIM;
  int dpart = (d < 32) ? (h*64 + d + 32) : (h*64 + d - 32);
  float q = qf[ib+dg], k = kf[ib+dg];
  float qp = qf[ib+dpart], kp = kf[ib+dpart];
  if (d < 32){ qp = -qp; kp = -kp; }
  size_t ob = (((size_t)b*NH + h)*SEQ + s)*HD + d;
  qh[ob] = (_Float16)(q*cs + qp*sn);
  kh[ob] = (_Float16)(k*cs + kp*sn);
}

// V repack+transpose: [B,S,D] f32 -> [B,H,HD,SP] f16, zero tail rows s>=SEQ
__global__ void tts_pack_vT(const float* __restrict__ vf, _Float16* __restrict__ vT){
  size_t i = (size_t)blockIdx.x*256 + threadIdx.x;
  size_t total = (size_t)BB*NH*HD*SP;
  if (i >= total) return;
  int s = (int)(i % SP); size_t r = i / SP;
  int d = (int)(r % HD); r /= HD;
  int h = (int)(r % NH); int b = (int)(r / NH);
  float v = (s < SEQ) ? vf[((size_t)b*SEQ + s)*DIM + h*HD + d] : 0.f;
  vT[i] = (_Float16)v;
}

// ---------------------------------------------------------------------------
// Flash-style causal attention. One wave = one 16-row query tile of one head.
// Processes 32 keys / iteration: full 16x32 P fragment, no zero-padded WMMA.
__global__ void __launch_bounds__(128)
tts_attn(const _Float16* __restrict__ qh, const _Float16* __restrict__ kh,
         const _Float16* __restrict__ vT, float* __restrict__ attn){
  const int lane = threadIdx.x, wv = threadIdx.y;
  const int qt = blockIdx.x*4 + wv;
  const int h = blockIdx.y, b = blockIdx.z;
  const int hg = lane >> 4, lr = lane & 15;
  const _Float16* Q  = qh + (((size_t)b*NH + h)*SEQ)*HD;
  const _Float16* Kb = kh + (((size_t)b*NH + h)*SEQ)*HD;
  const _Float16* Vt = vT + (((size_t)b*NH + h)*HD)*SP;
  __shared__ _Float16 plds[4][16][34];               // per-wave 16x32 P tile (+pad)

  v16h qa0, qa1;
  {
    const _Float16* Qr = Q + (size_t)(qt*16 + lr)*HD;
    v8h a0 = *(const v8h*)(Qr + hg*16);
    v8h a1 = *(const v8h*)(Qr + hg*16 + 8);
    v8h a2 = *(const v8h*)(Qr + 32 + hg*16);
    v8h a3 = *(const v8h*)(Qr + 40 + hg*16);
    #pragma unroll
    for (int e = 0; e < 8; ++e){
      qa0[e] = a0[e]; qa0[8+e] = a1[e];
      qa1[e] = a2[e]; qa1[8+e] = a3[e];
    }
  }
  v8f acc0 = {}, acc1 = {}, acc2 = {}, acc3 = {};
  float mrow[8], lrow[8];
  #pragma unroll
  for (int j = 0; j < 8; ++j){ mrow[j] = -1e30f; lrow[j] = 0.f; }

  for (int pr = 0; pr <= (qt >> 1); ++pr){
    const int kc = pr*32;                            // base key column
    const _Float16* Kr0 = Kb + (size_t)(kc + lr)*HD;
    const _Float16* Kr1 = Kb + (size_t)(kc + 16 + lr)*HD;
    v16h k00 = *(const v16h*)(Kr0 + hg*16);
    v16h k01 = *(const v16h*)(Kr0 + 32 + hg*16);
    v16h k10 = *(const v16h*)(Kr1 + hg*16);
    v16h k11 = *(const v16h*)(Kr1 + 32 + hg*16);
    v8f scA = {}, scB = {};
    scA = __builtin_amdgcn_wmma_f32_16x16x32_f16(false, qa0, false, k00, (short)0, scA, false, false);
    scA = __builtin_amdgcn_wmma_f32_16x16x32_f16(false, qa1, false, k01, (short)0, scA, false, false);
    scB = __builtin_amdgcn_wmma_f32_16x16x32_f16(false, qa0, false, k10, (short)0, scB, false, false);
    scB = __builtin_amdgcn_wmma_f32_16x16x32_f16(false, qa1, false, k11, (short)0, scB, false, false);
    #pragma unroll
    for (int j = 0; j < 8; ++j){
      int q = qt*16 + j + 8*hg;
      float vA = scA[j]*0.125f;                      // 1/sqrt(64)
      float vB = scB[j]*0.125f;
      if (kc + lr      > q) vA = -1e30f;             // causal mask (assignment: NaN-safe)
      if (kc + 16 + lr > q) vB = -1e30f;
      float m = fmaxf(vA, vB);
      m = fmaxf(m, __shfl_xor(m, 1));
      m = fmaxf(m, __shfl_xor(m, 2));
      m = fmaxf(m, __shfl_xor(m, 4));
      m = fmaxf(m, __shfl_xor(m, 8));
      float mnew  = fmaxf(mrow[j], m);
      float scale = __expf(mrow[j] - mnew);
      float pA = __expf(vA - mnew);
      float pB = __expf(vB - mnew);
      float ps = pA + pB;
      ps += __shfl_xor(ps, 1);
      ps += __shfl_xor(ps, 2);
      ps += __shfl_xor(ps, 4);
      ps += __shfl_xor(ps, 8);
      lrow[j] = lrow[j]*scale + ps;
      mrow[j] = mnew;
      acc0[j] *= scale; acc1[j] *= scale; acc2[j] *= scale; acc3[j] *= scale;
      plds[wv][j + 8*hg][lr]      = (_Float16)pA;    // P tile in [M][N32] order
      plds[wv][j + 8*hg][16 + lr] = (_Float16)pB;
    }
    asm volatile("s_wait_dscnt 0" ::: "memory");     // wave-local LDS ordering
    v16h pa;
    #pragma unroll
    for (int e = 0; e < 8; ++e){
      pa[e]   = plds[wv][lr][hg*16 + e];             // K = hg*16+e
      pa[8+e] = plds[wv][lr][hg*16 + 8 + e];         // K = hg*16+8+e
    }
    v16h vb0 = *(const v16h*)(Vt + (size_t)( 0 + lr)*SP + kc + hg*16);
    v16h vb1 = *(const v16h*)(Vt + (size_t)(16 + lr)*SP + kc + hg*16);
    v16h vb2 = *(const v16h*)(Vt + (size_t)(32 + lr)*SP + kc + hg*16);
    v16h vb3 = *(const v16h*)(Vt + (size_t)(48 + lr)*SP + kc + hg*16);
    acc0 = __builtin_amdgcn_wmma_f32_16x16x32_f16(false, pa, false, vb0, (short)0, acc0, false, false);
    acc1 = __builtin_amdgcn_wmma_f32_16x16x32_f16(false, pa, false, vb1, (short)0, acc1, false, false);
    acc2 = __builtin_amdgcn_wmma_f32_16x16x32_f16(false, pa, false, vb2, (short)0, acc2, false, false);
    acc3 = __builtin_amdgcn_wmma_f32_16x16x32_f16(false, pa, false, vb3, (short)0, acc3, false, false);
  }
  #pragma unroll
  for (int j = 0; j < 8; ++j){
    int m = qt*16 + j + 8*hg;
    size_t rb = ((size_t)b*SEQ + m)*DIM + (size_t)h*HD;
    float inv = 1.f / lrow[j];
    attn[rb +  0 + lr] = acc0[j]*inv;
    attn[rb + 16 + lr] = acc1[j]*inv;
    attn[rb + 32 + lr] = acc2[j]*inv;
    attn[rb + 48 + lr] = acc3[j]*inv;
  }
}

// ---------------------------------------------------------------------------
// ALIF spiking scan over S (sequential); writes h2 = x + mixed + spike
__global__ void tts_alif(const float* __restrict__ x, const float* __restrict__ mixed,
                         const float* __restrict__ bt, const float* __restrict__ asw,
                         float* __restrict__ h2){
  int b = blockIdx.y;
  int d = blockIdx.x*blockDim.x + threadIdx.x;
  float vmem = 0.f, ad = 0.f;
  float bth = bt[d], as = asw[d];
  size_t base = (size_t)b*SEQ*DIM + d;
  for (int t = 0; t < SEQ; ++t){
    size_t idx = base + (size_t)t*DIM;
    float it = mixed[idx];
    vmem = 0.99f*vmem + it;
    float th = bth + as*ad;
    float spike = sigmoidf_(25.f*(vmem - th));
    vmem -= spike*th;
    ad = 0.95f*ad + spike;
    h2[idx] = x[idx] + it + spike;
  }
}

// ---------------------------------------------------------------------------
// Tree level: raw = s[2i]+s[2i+1]; res=0.5*raw; n = rmsnorm(raw)*tn1
__global__ void tts_tree_pair(const float* __restrict__ s, const float* __restrict__ tn1,
                              float* __restrict__ nf, _Float16* __restrict__ nh,
                              float* __restrict__ res, int Sh){
  int b = blockIdx.y, i = blockIdx.x;
  const float* r0 = s + ((size_t)b*2*Sh + 2*i)*DIM;
  const float* r1 = r0 + DIM;
  size_t orow = ((size_t)b*Sh + i)*DIM;
  __shared__ float red[256];
  float ra0 = r0[threadIdx.x]       + r1[threadIdx.x];
  float ra1 = r0[threadIdx.x + 256] + r1[threadIdx.x + 256];
  red[threadIdx.x] = ra0*ra0 + ra1*ra1;
  __syncthreads();
  for (int st = 128; st > 0; st >>= 1){
    if (threadIdx.x < st) red[threadIdx.x] += red[threadIdx.x + st];
    __syncthreads();
  }
  float rn = rsqrtf(red[0]*(1.f/DIM) + EPSV);
  float n0 = ra0*rn*tn1[threadIdx.x];
  float n1 = ra1*rn*tn1[threadIdx.x + 256];
  nf[orow + threadIdx.x]       = n0;
  nf[orow + threadIdx.x + 256] = n1;
  nh[orow + threadIdx.x]       = (_Float16)n0;
  nh[orow + threadIdx.x + 256] = (_Float16)n1;
  res[orow + threadIdx.x]       = 0.5f*ra0;
  res[orow + threadIdx.x + 256] = 0.5f*ra1;
}

// s_new = rmsnorm(g*n + (1-g)*res)*tn2     (g already sigmoided by GEMM epilogue)
__global__ void tts_tree_blend(const float* __restrict__ g, const float* __restrict__ nf,
                               const float* __restrict__ res, const float* __restrict__ tn2,
                               float* __restrict__ sout){
  size_t base = (size_t)blockIdx.x*DIM;
  __shared__ float red[256];
  float v[2];
  #pragma unroll
  for (int t = 0; t < 2; ++t){
    int idx = threadIdx.x + t*256;
    float gg = g[base + idx];
    v[t] = gg*nf[base + idx] + (1.f - gg)*res[base + idx];
  }
  red[threadIdx.x] = v[0]*v[0] + v[1]*v[1];
  __syncthreads();
  for (int st = 128; st > 0; st >>= 1){
    if (threadIdx.x < st) red[threadIdx.x] += red[threadIdx.x + st];
    __syncthreads();
  }
  float rn = rsqrtf(red[0]*(1.f/DIM) + EPSV);
  #pragma unroll
  for (int t = 0; t < 2; ++t){
    int idx = threadIdx.x + t*256;
    sout[base + idx] = v[t]*rn*tn2[idx];
  }
}

__global__ void tts_add_root(const float* __restrict__ h2, const float* __restrict__ root,
                             float* __restrict__ h3){
  size_t i = (size_t)blockIdx.x*256 + threadIdx.x;
  if (i >= (size_t)BB*SEQ*DIM) return;
  int b = (int)(i / ((size_t)SEQ*DIM));
  int d = (int)(i % DIM);
  h3[i] = h2[i] + root[(size_t)b*DIM + d];
}

__global__ void tts_swiglu(const float* __restrict__ up, _Float16* __restrict__ acth){
  size_t i = (size_t)blockIdx.x*256 + threadIdx.x;
  if (i >= (size_t)BB*SEQ*DFF) return;
  size_t m = i / DFF; int j = (int)(i % DFF);
  float a  = up[m*(2*DFF) + j];
  float bb = up[m*(2*DFF) + DFF + j];
  acth[i] = (_Float16)(a * sigmoidf_(a) * bb);
}

// ---------------------------------------------------------------------------
extern "C" void kernel_launch(void* const* d_in, const int* in_sizes, int n_in,
                              void* d_out, int out_size, void* d_ws, size_t ws_size,
                              hipStream_t stream){
  (void)in_sizes; (void)n_in; (void)out_size; (void)ws_size;
  const float* x      = (const float*)d_in[0];
  const float* norm_w = (const float*)d_in[1];
  const float* wq     = (const float*)d_in[2];
  const float* wk     = (const float*)d_in[3];
  const float* wvv    = (const float*)d_in[4];
  const float* wo     = (const float*)d_in[5];
  const float* lat    = (const float*)d_in[6];
  const float* bt     = (const float*)d_in[7];
  const float* asw    = (const float*)d_in[8];
  const float* gate_w = (const float*)d_in[9];
  const float* tn1    = (const float*)d_in[10];
  const float* tn2    = (const float*)d_in[11];
  const float* ffn_nw = (const float*)d_in[12];
  const float* u_w    = (const float*)d_in[13];
  const float* v_w    = (const float*)d_in[14];
  const float* down_w = (const float*)d_in[15];
  float* out = (float*)d_out;

  const size_t TOK = (size_t)BB*SEQ;        // 4096 rows
  size_t off = 0;
  auto alloc = [&](size_t bytes)->void*{
    off = (off + 255) & ~(size_t)255;
    void* p = (char*)d_ws + off;
    off += bytes;
    return p;
  };
  // f16 buffers
  _Float16* h_h    = (_Float16*)alloc(TOK*DIM*2);
  _Float16* wq_h   = (_Float16*)alloc((size_t)DIM*DIM*2);
  _Float16* wk_h   = (_Float16*)alloc((size_t)DIM*DIM*2);
  _Float16* wv_h   = (_Float16*)alloc((size_t)DIM*DIM*2);
  _Float16* wo_h   = (_Float16*)alloc((size_t)DIM*DIM*2);
  _Float16* gate_h = (_Float16*)alloc((size_t)DIM*DIM*2);
  _Float16* lat_h  = (_Float16*)alloc((size_t)SEQ*SEQ*2);
  _Float16* uT_h   = (_Float16*)alloc((size_t)RANK*DIM*2);
  _Float16* vT2_h  = (_Float16*)alloc((size_t)2*DFF*RANK*2);
  _Float16* dnT_h  = (_Float16*)alloc((size_t)DIM*DFF*2);
  _Float16* q_h    = (_Float16*)alloc(TOK*DIM*2);
  _Float16* k_h    = (_Float16*)alloc(TOK*DIM*2);
  _Float16* vT_h   = (_Float16*)alloc((size_t)BB*NH*HD*SP*2);
  _Float16* attn_h = (_Float16*)alloc(TOK*DIM*2);
  _Float16* apT_h  = (_Float16*)alloc((size_t)BB*DIM*SEQ*2);
  _Float16* n_h    = (_Float16*)alloc((size_t)BB*(SEQ/2)*DIM*2);
  _Float16* hn_h   = (_Float16*)alloc(TOK*DIM*2);
  _Float16* t1_h   = (_Float16*)alloc(TOK*RANK*2);
  _Float16* act_h  = (_Float16*)alloc(TOK*DFF*2);
  // f32 buffers
  float* qf    = (float*)alloc(TOK*DIM*4);
  float* kf    = (float*)alloc(TOK*DIM*4);
  float* vf    = (float*)alloc(TOK*DIM*4);
  float* attnf = (float*)alloc(TOK*DIM*4);
  float* aproj = (float*)alloc(TOK*DIM*4);
  float* mixed = (float*)alloc(TOK*DIM*4);
  float* h2    = (float*)alloc(TOK*DIM*4);
  float* nf    = (float*)alloc((size_t)BB*(SEQ/2)*DIM*4);
  float* resf  = (float*)alloc((size_t)BB*(SEQ/2)*DIM*4);
  float* gpre  = (float*)alloc((size_t)BB*(SEQ/2)*DIM*4);
  float* sA    = (float*)alloc((size_t)BB*(SEQ/2)*DIM*4);
  float* sB    = (float*)alloc((size_t)BB*(SEQ/2)*DIM*4);
  float* h3    = (float*)alloc(TOK*DIM*4);
  float* upf   = (float*)alloc(TOK*(size_t)(2*DFF)*4);

  auto gemm = [&](const _Float16* A, const _Float16* Bt, float* C, _Float16* Ch,
                  const float* addsrc, int M, int N, int K, int act){
    dim3 blk(32, 8, 1);
    dim3 grd((unsigned)((N/32 + 7)/8), (unsigned)((M + 31)/32), 1);
    tts_gemm_nt<<<grd, blk, 0, stream>>>(A, Bt, C, Ch, addsrc, M, N, K, act);
  };

  // ---- weight conversions (deterministic, redone every call) ----
  const size_t WSZ = (size_t)DIM*DIM;
  tts_f2h<<<(unsigned)((WSZ+255)/256), 256, 0, stream>>>(wq, wq_h, WSZ);
  tts_f2h<<<(unsigned)((WSZ+255)/256), 256, 0, stream>>>(wk, wk_h, WSZ);
  tts_f2h<<<(unsigned)((WSZ+255)/256), 256, 0, stream>>>(wvv, wv_h, WSZ);
  tts_f2h<<<(unsigned)((WSZ+255)/256), 256, 0, stream>>>(wo, wo_h, WSZ);
  tts_f2h<<<(unsigned)((WSZ+255)/256), 256, 0, stream>>>(gate_w, gate_h, WSZ);
  tts_tril_f2h<<<(unsigned)(((size_t)SEQ*SEQ+255)/256), 256, 0, stream>>>(lat, lat_h);
  tts_transpose_f2h<<<dim3((DIM*RANK+255)/256, 1), 256, 0, stream>>>(u_w, uT_h, DIM, RANK);
  tts_transpose_f2h<<<dim3((RANK*2*DFF+255)/256, 1), 256, 0, stream>>>(v_w, vT2_h, RANK, 2*DFF);
  tts_transpose_f2h<<<dim3((DFF*DIM+255)/256, 1), 256, 0, stream>>>(down_w, dnT_h, DFF, DIM);

  // ---- pre-norm + QKV projections ----
  tts_rmsnorm<<<(unsigned)TOK, 256, 0, stream>>>(x, norm_w, nullptr, h_h);
  gemm(h_h, wq_h, qf, nullptr, nullptr, (int)TOK, DIM, DIM, 0);
  gemm(h_h, wk_h, kf, nullptr, nullptr, (int)TOK, DIM, DIM, 0);
  gemm(h_h, wv_h, vf, nullptr, nullptr, (int)TOK, DIM, DIM, 0);

  // ---- RoPE + head packing ----
  tts_rope_pack<<<(unsigned)TOK, 512, 0, stream>>>(qf, kf, q_h, k_h);
  {
    size_t tot = (size_t)BB*NH*HD*SP;
    tts_pack_vT<<<(unsigned)((tot+255)/256), 256, 0, stream>>>(vf, vT_h);
  }

  // ---- causal flash attention ----
  tts_attn<<<dim3(SEQ/16/4, NH, BB), dim3(32, 4, 1), 0, stream>>>(q_h, k_h, vT_h, attnf);

  // ---- output projection + lateral mixing ----
  tts_f2h<<<(unsigned)((TOK*DIM+255)/256), 256, 0, stream>>>(attnf, attn_h, TOK*DIM);
  gemm(attn_h, wo_h, aproj, nullptr, nullptr, (int)TOK, DIM, DIM, 0);
  tts_transpose_f2h<<<dim3((SEQ*DIM+255)/256, BB), 256, 0, stream>>>(aproj, apT_h, SEQ, DIM);
  for (int b = 0; b < BB; ++b)
    gemm(lat_h, apT_h + (size_t)b*DIM*SEQ, mixed + (size_t)b*SEQ*DIM,
         nullptr, nullptr, SEQ, DIM, SEQ, 0);

  // ---- ALIF scan + residual (h2 = x + mixed + spikes) ----
  tts_alif<<<dim3(DIM/256, BB), 256, 0, stream>>>(x, mixed, bt, asw, h2);

  // ---- binary-tree recursion ----
  const float* scur = h2;
  int Sl = SEQ;
  for (int l = 0; l < 11; ++l){
    int Sh = Sl >> 1;
    float* sout = (l & 1) ? sA : sB;
    tts_tree_pair<<<dim3(Sh, BB), 256, 0, stream>>>(scur, tn1, nf, n_h, resf, Sh);
    gemm(n_h, gate_h, gpre, nullptr, nullptr, BB*Sh, DIM, DIM, 1);
    tts_tree_blend<<<(unsigned)(BB*Sh), 256, 0, stream>>>(gpre, nf, resf, tn2, sout);
    scur = sout; Sl = Sh;
  }
  tts_add_root<<<(unsigned)((TOK*DIM+255)/256), 256, 0, stream>>>(h2, scur, h3);

  // ---- low-rank SwiGLU FFN, residual fused into final GEMM ----
  tts_rmsnorm<<<(unsigned)TOK, 256, 0, stream>>>(h3, ffn_nw, nullptr, hn_h);
  gemm(hn_h, uT_h, nullptr, t1_h, nullptr, (int)TOK, RANK, DIM, 0);
  gemm(t1_h, vT2_h, upf, nullptr, nullptr, (int)TOK, 2*DFF, RANK, 0);
  tts_swiglu<<<(unsigned)((TOK*DFF+255)/256), 256, 0, stream>>>(upf, act_h);
  gemm(act_h, dnT_h, out, nullptr, h3, (int)TOK, DIM, DFF, 0);
}
